// Encoder_8117488189536
// MI455X (gfx1250) — matrix operand order repaired and checked by reading, not compiled, not run on hardware
//
#include <hip/hip_runtime.h>

#define BSZ    128
#define CCH    32
#define HH     128
#define WW     64
#define DD     2048            // C*W
#define NNODES (BSZ*HH)        // 16384
#define KNN    8
#define NLAYERS 4
#define EPSBN  1e-5f

typedef __attribute__((ext_vector_type(16))) __bf16 v16bf;
typedef __attribute__((ext_vector_type(8)))  __bf16 v8bf;
typedef __attribute__((ext_vector_type(8)))  float  v8f;

static __device__ __forceinline__ unsigned short f2bf(float f) {
    unsigned u = __float_as_uint(f);
    u += 0x7FFFu + ((u >> 16) & 1u);          // round to nearest even
    return (unsigned short)(u >> 16);
}

// async copy 16B global -> LDS (ASYNCcnt-tracked); lds = raw 32-bit LDS offset
static __device__ __forceinline__ void async_cp16(const unsigned short* g, unsigned lds) {
    unsigned long long ga = (unsigned long long)(uintptr_t)g;
    asm volatile("global_load_async_to_lds_b128 %0, %1, off"
                 :: "v"(lds), "v"(ga) : "memory");
}

// ---------------- rearrange: x[b][c][h][w] -> nodes[(b h)][(c w)] (f32 + bf16) ----------
__global__ void k_rearrange(const float* __restrict__ x,
                            float* __restrict__ xf, unsigned short* __restrict__ xb) {
    int tid = blockIdx.x * blockDim.x + threadIdx.x;       // exact grid: NNODES*DD
    int d = tid % DD, n = tid / DD;
    int c = d / WW,  w = d % WW;
    int b = n / HH,  h = n % HH;
    float v = x[(((size_t)b * CCH + c) * HH + h) * WW + w];
    xf[tid] = v;
    xb[tid] = f2bf(v);
}

// ---------------- per-node L2 normalize -> bf16 (for cosine kNN) ------------------------
__global__ void k_normalize(const float* __restrict__ xf, unsigned short* __restrict__ xn) {
    __shared__ float red[256];
    int n = blockIdx.x, t = threadIdx.x;
    const float* row = xf + (size_t)n * DD;
    float vals[8]; float ss = 0.f;
#pragma unroll
    for (int i = 0; i < 8; ++i) { float v = row[t + 256 * i]; vals[i] = v; ss += v * v; }
    red[t] = ss; __syncthreads();
    for (int s = 128; s > 0; s >>= 1) { if (t < s) red[t] += red[t + s]; __syncthreads(); }
    float scale = 1.f / (sqrtf(red[0]) + 1e-12f);
    unsigned short* orow = xn + (size_t)n * DD;
#pragma unroll
    for (int i = 0; i < 8; ++i) orow[t + 256 * i] = f2bf(vals[i] * scale);
}

// ---------------- per-batch cosine similarity via WMMA: sim[b] = xn_b @ xn_b^T ----------
__global__ void __launch_bounds__(256) k_sim(const unsigned short* __restrict__ xn,
                                             float* __restrict__ sim) {
    int b = blockIdx.x;
    int wave = threadIdx.x >> 5, lane = threadIdx.x & 31;
    int lm = lane & 15, lh = lane >> 4;
    const unsigned short* base = xn + (size_t)b * HH * DD;
    const unsigned short* arow = base + (size_t)(wave * 16 + lm) * DD;
    v8f acc[8] = {};
    for (int k0 = 0; k0 < DD; k0 += 32) {
        v16bf a = *(const v16bf*)(arow + k0 + lh * 16);
#pragma unroll
        for (int nt = 0; nt < 8; ++nt) {
            v16bf bf = *(const v16bf*)(base + (size_t)(nt * 16 + lm) * DD + k0 + lh * 16);
            acc[nt] = __builtin_amdgcn_wmma_f32_16x16x32_bf16(
                false, a, false, bf, (short)0, acc[nt], false, false);
        }
    }
#pragma unroll
    for (int nt = 0; nt < 8; ++nt) {
        int col = nt * 16 + lm;
#pragma unroll
        for (int i = 0; i < 8; ++i) {
            int m = wave * 16 + lh * 8 + i;
            sim[((size_t)b * HH + m) * HH + col] = acc[nt][i];
        }
    }
}

// ---------------- top-K neighbors (descending, first-index ties, self excluded) ---------
__global__ void k_topk(const float* __restrict__ sim, int* __restrict__ srcIdx) {
    int n = blockIdx.x * blockDim.x + threadIdx.x;          // exact grid: NNODES
    int b = n / HH, r = n % HH;
    const float* s = sim + (size_t)n * HH;
    int chosen[KNN];
    for (int k = 0; k < KNN; ++k) {
        float best = -1e30f; int bi = 0;
        for (int j = 0; j < HH; ++j) {
            if (j == r) continue;
            bool used = false;
            for (int q = 0; q < k; ++q) used |= (chosen[q] == j);
            if (used) continue;
            float v = s[j];
            if (v > best) { best = v; bi = j; }
        }
        chosen[k] = bi;
        srcIdx[n * KNN + k] = b * HH + bi;
    }
}

// ---------------- weight transpose + bf16 cast: w[fin][fout] -> wt[fout][fin] -----------
__global__ void k_wt(const float* __restrict__ w, unsigned short* __restrict__ wt,
                     int fin, int fout) {
    int tid = blockIdx.x * blockDim.x + threadIdx.x;
    if (tid >= fin * fout) return;
    int i = tid / fout, o = tid % fout;
    wt[(size_t)o * fin + i] = f2bf(w[tid]);
}

// ---------------- fused dual WMMA GEMM with async-LDS double-buffered B -----------------
// BtPair = [2*fout][fin] bf16 (gwT rows then rwT rows, contiguous).
// grid (N/128, 2*fout/64); first half of blockIdx.y writes C0 (hlin), second half C1 (res)
#define BPAD 8
__global__ void __launch_bounds__(256) k_gemm2(const unsigned short* __restrict__ A,
                                               const unsigned short* __restrict__ BtPair,
                                               float* __restrict__ C0, float* __restrict__ C1,
                                               int Kd, int fout) {
    __shared__ unsigned short lb[2][64][64 + BPAD];         // 2 x 9.2 KB B tiles
    int wave = threadIdx.x >> 5, lane = threadIdx.x & 31;
    int lm = lane & 15, lh = lane >> 4;
    int nyhalf = gridDim.y >> 1;
    int half = (blockIdx.y >= nyhalf) ? 1 : 0;
    float* Cm = half ? C1 : C0;
    int colBase = (blockIdx.y - half * nyhalf) * 64;

    // fill assignment: thread t copies 32B: Bt row (btRow), k-seg of 16 bf16 x2
    int fcol = threadIdx.x >> 2;                            // 0..63
    int fseg = threadIdx.x & 3;                             // 0..3
    const unsigned short* fgp =
        BtPair + (size_t)(blockIdx.y * 64 + fcol) * Kd + fseg * 16;
    unsigned flds0 = (unsigned)(uintptr_t)&lb[0][fcol][fseg * 16];
    unsigned flds1 = (unsigned)(uintptr_t)&lb[1][fcol][fseg * 16];

    const unsigned short* arow = A + (size_t)(blockIdx.x * 128 + wave * 16 + lm) * Kd;
    v8f acc[4] = {};

    // prologue: stage chunk 0 into buffer 0
    async_cp16(fgp, flds0);
    async_cp16(fgp + 8, flds0 + 16);

    int buf = 0;
    for (int kc = 0; kc < Kd; kc += 64) {
        if (kc + 64 < Kd) {                                 // stream next chunk into other buf
            const unsigned short* g = fgp + kc + 64;
            unsigned l = buf ? flds0 : flds1;
            async_cp16(g, l);
            async_cp16(g + 8, l + 16);
            asm volatile("s_wait_asynccnt 2" ::: "memory"); // current buf's fill done
        } else {
            asm volatile("s_wait_asynccnt 0" ::: "memory");
        }
        __syncthreads();                                    // tile visible to all waves

        // A fragments for both k-steps of this chunk (issued together -> pipelined)
        v16bf a0 = *(const v16bf*)(arow + kc + lh * 16);
        v16bf a1 = *(const v16bf*)(arow + kc + 32 + lh * 16);
#pragma unroll
        for (int nt = 0; nt < 4; ++nt) {
            v16bf bf = *(const v16bf*)&lb[buf][nt * 16 + lm][lh * 16];
            acc[nt] = __builtin_amdgcn_wmma_f32_16x16x32_bf16(
                false, a0, false, bf, (short)0, acc[nt], false, false);
        }
#pragma unroll
        for (int nt = 0; nt < 4; ++nt) {
            v16bf bf = *(const v16bf*)&lb[buf][nt * 16 + lm][32 + lh * 16];
            acc[nt] = __builtin_amdgcn_wmma_f32_16x16x32_bf16(
                false, a1, false, bf, (short)0, acc[nt], false, false);
        }
        __syncthreads();                                    // done reading buf before refill
        buf ^= 1;
    }

    int rowTop = blockIdx.x * 128 + wave * 16 + lh * 8;
#pragma unroll
    for (int nt = 0; nt < 4; ++nt) {
        int col = colBase + nt * 16 + lm;
#pragma unroll
        for (int i = 0; i < 8; ++i)
            Cm[(size_t)(rowTop + i) * fout + col] = acc[nt][i];
    }
}

// ---------------- aggregate (deg==9 => norm==1/9) + gb + ReLU + grouped 4x4 conv --------
__global__ void k_agg(const float* __restrict__ hlin, const int* __restrict__ srcIdx,
                      const float* __restrict__ gb, const float* __restrict__ cw,
                      const float* __restrict__ cb, float* __restrict__ y, int fout) {
    int G = fout >> 2;
    int tid = blockIdx.x * blockDim.x + threadIdx.x;        // exact grid: NNODES*G
    int n = tid / G, g = tid % G;
    const float* hp = hlin + (size_t)n * fout + g * 4;
    float s0 = hp[0], s1 = hp[1], s2 = hp[2], s3 = hp[3];
#pragma unroll
    for (int k = 0; k < KNN; ++k) {
        int j = srcIdx[n * KNN + k];
        const float* q = hlin + (size_t)j * fout + g * 4;
        s0 += q[0]; s1 += q[1]; s2 += q[2]; s3 += q[3];
    }
    const float inv9 = 1.f / 9.f;
    float t0 = fmaxf(s0 * inv9 + gb[g * 4 + 0], 0.f);
    float t1 = fmaxf(s1 * inv9 + gb[g * 4 + 1], 0.f);
    float t2 = fmaxf(s2 * inv9 + gb[g * 4 + 2], 0.f);
    float t3 = fmaxf(s3 * inv9 + gb[g * 4 + 3], 0.f);
    float* yo = y + (size_t)n * fout + g * 4;
#pragma unroll
    for (int o = 0; o < 4; ++o) {                           // y[n,g,o] = sum_i t_i*cw[g,o,i]
        const float* c4 = cw + (size_t)(g * 4 + o) * 4;
        yo[o] = t0 * c4[0] + t1 * c4[1] + t2 * c4[2] + t3 * c4[3] + cb[g * 4 + o];
    }
}

// ---------------- zero small stats buffer -----------------------------------------------
__global__ void k_zero(float* __restrict__ p, int ncount) {
    int t = blockIdx.x * blockDim.x + threadIdx.x;
    if (t < ncount) p[t] = 0.f;
}

// ---------------- BN stats: coalesced tiled reduction over nodes ------------------------
__global__ void k_bnred(const float* __restrict__ y, float* __restrict__ bsum,
                        float* __restrict__ bsq, int fout) {
    __shared__ float ssum[256], ssq[256];
    int t = threadIdx.x;
    int c = blockIdx.x * 64 + (t & 63);
    int rl = t >> 6;                                        // 0..3
    int chunk = NNODES / gridDim.y;
    int n0 = blockIdx.y * chunk;
    float s = 0.f, q = 0.f;
    for (int n = n0 + rl; n < n0 + chunk; n += 4) {
        float v = y[(size_t)n * fout + c];
        s += v; q += v * v;
    }
    ssum[t] = s; ssq[t] = q; __syncthreads();
    if (rl == 0) {
        s = ssum[t] + ssum[t + 64] + ssum[t + 128] + ssum[t + 192];
        q = ssq[t]  + ssq[t + 64]  + ssq[t + 128]  + ssq[t + 192];
        atomicAdd(&bsum[c], s);
        atomicAdd(&bsq[c], q);
    }
}

// ---------------- BN apply + affine + residual (+rb), write f32 (+bf16 for next layer) --
__global__ void k_bnapply(const float* __restrict__ y, const float* __restrict__ res,
                          const float* __restrict__ bsum, const float* __restrict__ bsq,
                          const float* __restrict__ bng, const float* __restrict__ bnb,
                          const float* __restrict__ rb,
                          float* __restrict__ outf, unsigned short* __restrict__ outb,
                          int fout, int writeBf) {
    int tid = blockIdx.x * blockDim.x + threadIdx.x;        // exact grid: NNODES*fout
    int c = tid % fout;
    const float Ninv = 1.f / (float)NNODES;
    float mu  = bsum[c] * Ninv;
    float var = bsq[c] * Ninv - mu * mu;
    float v = (y[tid] - mu) * rsqrtf(var + EPSBN) * bng[c] + bnb[c] + res[tid] + rb[c];
    outf[tid] = v;
    if (writeBf) outb[tid] = f2bf(v);
}

// =========================================================================================
extern "C" void kernel_launch(void* const* d_in, const int* in_sizes, int n_in,
                              void* d_out, int out_size, void* d_ws, size_t ws_size,
                              hipStream_t stream) {
    (void)in_sizes; (void)n_in; (void)out_size; (void)ws_size;
    const float* x = (const float*)d_in[0];
    const float *gw[NLAYERS], *gb[NLAYERS], *cw[NLAYERS], *cb[NLAYERS],
                *bng[NLAYERS], *bnb[NLAYERS], *rw[NLAYERS], *rb[NLAYERS];
    for (int i = 0; i < NLAYERS; ++i) {
        gw[i]  = (const float*)d_in[1 + 8 * i + 0];
        gb[i]  = (const float*)d_in[1 + 8 * i + 1];
        cw[i]  = (const float*)d_in[1 + 8 * i + 2];
        cb[i]  = (const float*)d_in[1 + 8 * i + 3];
        bng[i] = (const float*)d_in[1 + 8 * i + 4];
        bnb[i] = (const float*)d_in[1 + 8 * i + 5];
        rw[i]  = (const float*)d_in[1 + 8 * i + 6];
        rb[i]  = (const float*)d_in[1 + 8 * i + 7];
    }
    const int fins[NLAYERS]  = {2048, 1024, 512, 256};
    const int fouts[NLAYERS] = {1024, 512, 256, 128};

    // ---- workspace carve (256B aligned) ----
    char* p = (char*)d_ws;
    auto carve = [&](size_t bytes) -> char* {
        char* r = p; p += (bytes + 255) & ~(size_t)255; return r;
    };
    float*          xf   = (float*)carve((size_t)NNODES * DD * 4);          // 134 MB
    unsigned short* xb   = (unsigned short*)carve((size_t)NNODES * DD * 2); //  67 MB
    float*          hlin = (float*)carve((size_t)NNODES * 1024 * 4);        //  67 MB
    unsigned short* xn   = (unsigned short*)hlin;  // alias: xn dead before first GEMM
    float*          res  = (float*)carve((size_t)NNODES * 1024 * 4);        //  67 MB
    float*          ybuf = (float*)carve((size_t)NNODES * 1024 * 4);        //  67 MB
    float*          sim  = ybuf;                   // alias: sim dead after top-k
    int*            srcI = (int*)carve((size_t)NNODES * KNN * 4);
    float*          bsum = (float*)carve(2048 * 4);
    float*          bsq  = (float*)carve(2048 * 4);
    size_t wtot = 0;
    for (int i = 0; i < NLAYERS; ++i) wtot += 2 * (size_t)fins[i] * fouts[i];
    unsigned short* wT = (unsigned short*)carve(wtot * 2);                  //  11 MB
    unsigned short *gwT[NLAYERS], *rwT[NLAYERS];
    {
        size_t off = 0;
        for (int i = 0; i < NLAYERS; ++i) {
            gwT[i] = wT + off; off += (size_t)fins[i] * fouts[i];   // gwT then rwT:
            rwT[i] = wT + off; off += (size_t)fins[i] * fouts[i];   // contiguous pair
        }
    }

    // ---- weight prep (bf16 transposed) ----
    for (int i = 0; i < NLAYERS; ++i) {
        int ne = fins[i] * fouts[i];
        k_wt<<<(ne + 255) / 256, 256, 0, stream>>>(gw[i], gwT[i], fins[i], fouts[i]);
        k_wt<<<(ne + 255) / 256, 256, 0, stream>>>(rw[i], rwT[i], fins[i], fouts[i]);
    }

    // ---- node prep + kNN graph ----
    k_rearrange<<<(size_t)NNODES * DD / 256, 256, 0, stream>>>(x, xf, xb);
    k_normalize<<<NNODES, 256, 0, stream>>>(xf, xn);
    k_sim<<<BSZ, 256, 0, stream>>>(xn, sim);
    k_topk<<<NNODES / 256, 256, 0, stream>>>(sim, srcI);

    // ---- layers ----
    for (int i = 0; i < NLAYERS; ++i) {
        int fin = fins[i], fout = fouts[i];
        dim3 gg(NNODES / 128, (2 * fout) / 64);
        k_gemm2<<<gg, 256, 0, stream>>>(xb, gwT[i], hlin, res, fin, fout);
        k_agg<<<(size_t)NNODES * (fout / 4) / 256, 256, 0, stream>>>(
            hlin, srcI, gb[i], cw[i], cb[i], ybuf, fout);
        k_zero<<<16, 256, 0, stream>>>(bsum, 4096);         // bsum+bsq contiguous
        k_bnred<<<dim3(fout / 64, 32), 256, 0, stream>>>(ybuf, bsum, bsq, fout);
        int last = (i == NLAYERS - 1);
        k_bnapply<<<(size_t)NNODES * fout / 256, 256, 0, stream>>>(
            ybuf, res, bsum, bsq, bng[i], bnb[i], rb[i],
            last ? (float*)d_out : xf, xb, fout, last ? 0 : 1);
    }
}
